// Convolution_18296560681267
// MI455X (gfx1250) — matrix operand order repaired and checked by reading, not compiled, not run on hardware
//
#include <hip/hip_runtime.h>
#include <math.h>

// ---------------------------------------------------------------------------
// e3nn-style equivariant graph convolution, fused for MI455X (gfx1250).
// GEMMs on v_wmma_f32_16x16x32_f16; weights pre-swizzled to B-fragment layout
// (one v16h load per fragment); A-tiles and all outputs staged through LDS so
// global traffic is coalesced float4. Fragment loops kept rolled (unroll 1)
// to bound register pressure -> no scratch spills.
// ---------------------------------------------------------------------------

typedef __attribute__((ext_vector_type(16))) _Float16 v16h;
typedef __attribute__((ext_vector_type(8)))  float    v8f;

#define MUL 32
#define NEF 16
#define RH  64
#define WNUM 160
#define C_S 0.3826834323650898f
#define C_X 0.9238795325112867f
#define INV_SQRT3 0.5773502691896258f
#define INV_SQRT2 0.7071067811865475f
#define INV_NN 0.25f                  // 1/sqrt(16 neighbors)
#define INV_S32 0.17677669529663687f  // 1/sqrt(32)
#define INV_S16 0.25f                 // 1/sqrt(16)
#define INV_S64 0.125f                // 1/sqrt(64)
#define INV_S96 0.10206207261596575f  // 1/sqrt(96)

// Pre-swizzled weight-fragment table (each fragment: 32 lanes x 16 f16 = 512)
#define FR_SC0 0    // sc_w0   (2)
#define FR_L10 2    // lin1_w0 (2)
#define FR_SC1 4    // sc_w1   (2)
#define FR_L11 6    // lin1_w1 (2)
#define FR_FC1 8    // fc_w1 padded K16->32 (4)
#define FR_FC2 12   // fc_w2, kt*10+nt (20)
#define FR_L20 32   // lin2_w0, kt*2+nt (4)
#define FR_L21 36   // lin2_w1, kt*2+nt (6)
#define FR_TOTAL 42

// ---- WMMA per-lane layout helpers (wave32, 16x16x32 f16 -> f32) ----
__device__ __forceinline__ int a_k(int lane, int i) {
    return ((i < 8) ? i : (i + 8)) + ((lane >> 4) << 3);
}
__device__ __forceinline__ int b_k(int lane, int i) {
    return ((i < 8) ? i : (i + 8)) + ((lane >> 4) << 3);
}
__device__ __forceinline__ int d_row(int lane, int r) {
    return r + ((lane >> 4) << 3);
}

__device__ __forceinline__ v8f wmma16(v16h a, v16h b, v8f c) {
    return __builtin_amdgcn_wmma_f32_16x16x32_f16(
        false, a, false, b, (short)0, c, false, false);
}

__device__ __forceinline__ v16h ldfrag(const _Float16* __restrict__ wf, int f, int lane) {
    return *(const v16h*)(wf + (size_t)f * 512 + lane * 16);
}

// pack 8 contiguous floats (two float4) into a[o..o+7] as f16
__device__ __forceinline__ void cvt8(v16h& a, int o, const float4 p, const float4 q) {
    a[o + 0] = (_Float16)p.x; a[o + 1] = (_Float16)p.y;
    a[o + 2] = (_Float16)p.z; a[o + 3] = (_Float16)p.w;
    a[o + 4] = (_Float16)q.x; a[o + 5] = (_Float16)q.y;
    a[o + 6] = (_Float16)q.z; a[o + 7] = (_Float16)q.w;
}

__device__ __forceinline__ float silu(float x) {
    return x / (1.0f + __expf(-x));
}

// ---------------------------------------------------------------------------
// Kernel P: swizzle all weights into WMMA B-fragment layout (f16), zero-padded.
// ---------------------------------------------------------------------------
__global__ void prep_weights(const float* __restrict__ sc_w0, const float* __restrict__ sc_w1,
                             const float* __restrict__ l1_w0, const float* __restrict__ l1_w1,
                             const float* __restrict__ fc_w1, const float* __restrict__ fc_w2,
                             const float* __restrict__ l2_w0, const float* __restrict__ l2_w1,
                             _Float16* __restrict__ wf)
{
    const int wave = threadIdx.x >> 5, lane = threadIdx.x & 31;
    for (int f = wave; f < FR_TOTAL; f += 8) {
        const float* W; int kt = 0, nt, ncols, kmax;
        if (f < 2)       { W = sc_w0; nt = f;      ncols = 32;  kmax = 32; }
        else if (f < 4)  { W = l1_w0; nt = f - 2;  ncols = 32;  kmax = 32; }
        else if (f < 6)  { W = sc_w1; nt = f - 4;  ncols = 32;  kmax = 32; }
        else if (f < 8)  { W = l1_w1; nt = f - 6;  ncols = 32;  kmax = 32; }
        else if (f < 12) { W = fc_w1; nt = f - 8;  ncols = 64;  kmax = 16; }
        else if (f < 32) { int ff = f - 12; W = fc_w2; kt = ff / 10; nt = ff % 10; ncols = 160; kmax = 64; }
        else if (f < 36) { int ff = f - 32; W = l2_w0; kt = ff >> 1; nt = ff & 1; ncols = 32; kmax = 64; }
        else             { int ff = f - 36; W = l2_w1; kt = ff >> 1; nt = ff & 1; ncols = 32; kmax = 96; }
        v16h b;
#pragma unroll
        for (int i = 0; i < 16; ++i) {
            const int k = kt * 32 + b_k(lane, i);
            b[i] = (k < kmax) ? (_Float16)W[k * ncols + nt * 16 + (lane & 15)]
                              : (_Float16)0.0f;
        }
        *(v16h*)(wf + (size_t)f * 512 + lane * 16) = b;
    }
}

// ---------------------------------------------------------------------------
// Kernel 0: zero g0/g1 accumulators, vectorized (count must be multiple of 4)
// ---------------------------------------------------------------------------
__global__ void zero_g(float4* __restrict__ g, long long count4) {
    long long i = (long long)blockIdx.x * blockDim.x + threadIdx.x;
    const long long stride = (long long)gridDim.x * blockDim.x;
    const float4 zero = {0.f, 0.f, 0.f, 0.f};
    for (; i < count4; i += stride) g[i] = zero;
}

// ---------------------------------------------------------------------------
// Kernel 1: per-node fctp for sc and lin1. One wave = 16 nodes, 16 WMMAs.
//   C_S*s -> staged in LDS, coalesced float4 stores into d_out
//   z     -> staged in LDS, coalesced float4 stores into zpack
//            zpack[n][u] = {z0, z1x, z1y, z1z}
// ---------------------------------------------------------------------------
__global__ void __launch_bounds__(32, 1)
node_lin(const float* __restrict__ node_input,
         const float* __restrict__ node_attr,
         const _Float16* __restrict__ wf,
         float* __restrict__ zpack,
         float* __restrict__ out, int n)
{
    __shared__ __align__(16) float xs[16 * 128];   // input tile
    __shared__ __align__(16) float ot[16 * 128];   // C_S*s tile
    __shared__ __align__(16) float zt[16 * 128];   // z tile
    __shared__ float av[16];
    const int lane = threadIdx.x & 31;
    int base = blockIdx.x * 16;
    if (base + 16 > n) base = n - 16;

#pragma unroll
    for (int j = 0; j < 16; ++j) {
        const int idx = j * 32 + lane;
        const int row = idx >> 5, c4 = (idx & 31) * 4;
        *(float4*)&xs[row * 128 + c4] =
            *(const float4*)&node_input[(size_t)(base + row) * 128 + c4];
    }
    if (lane < 16) av[lane] = node_attr[base + lane];
    __syncthreads();

    float afac[8];
#pragma unroll
    for (int r = 0; r < 8; ++r) afac[r] = av[d_row(lane, r)] * INV_S32;

    const int r0 = (lane & 15) * 128;
    const int kb = (lane >> 4) << 3;
    const int col0 = lane & 15;

    // ---- scalar channel ----
    v16h ax0;
    cvt8(ax0, 0, *(const float4*)&xs[r0 + kb],      *(const float4*)&xs[r0 + kb + 4]);
    cvt8(ax0, 8, *(const float4*)&xs[r0 + 16 + kb], *(const float4*)&xs[r0 + 16 + kb + 4]);

#pragma unroll 1
    for (int half = 0; half < 2; ++half) {
        const int col = half * 16 + col0;
        v8f cs = {};
        cs = wmma16(ax0, ldfrag(wf, FR_SC0 + half, lane), cs);
#pragma unroll
        for (int r = 0; r < 8; ++r)
            ot[d_row(lane, r) * 128 + col] = C_S * cs[r] * afac[r];
        v8f cz = {};
        cz = wmma16(ax0, ldfrag(wf, FR_L10 + half, lane), cz);
#pragma unroll
        for (int r = 0; r < 8; ++r)
            zt[d_row(lane, r) * 128 + col * 4] = cz[r] * afac[r];
    }

    // ---- vector channel, per m (rolled: bounds register pressure) ----
#pragma unroll 1
    for (int m = 0; m < 3; ++m) {
        v16h ax1;
#pragma unroll
        for (int i = 0; i < 16; ++i)
            ax1[i] = (_Float16)xs[r0 + 32 + a_k(lane, i) * 3 + m];
#pragma unroll 1
        for (int half = 0; half < 2; ++half) {
            const int col = half * 16 + col0;
            v8f cs = {};
            cs = wmma16(ax1, ldfrag(wf, FR_SC1 + half, lane), cs);
#pragma unroll
            for (int r = 0; r < 8; ++r)
                ot[d_row(lane, r) * 128 + 32 + col * 3 + m] = C_S * cs[r] * afac[r];
            v8f cz = {};
            cz = wmma16(ax1, ldfrag(wf, FR_L11 + half, lane), cz);
#pragma unroll
            for (int r = 0; r < 8; ++r)
                zt[d_row(lane, r) * 128 + col * 4 + 1 + m] = cz[r] * afac[r];
        }
    }
    __syncthreads();

    // coalesced vectorized writeback of both tiles
#pragma unroll
    for (int j = 0; j < 16; ++j) {
        const int idx = j * 32 + lane;
        const int row = idx >> 5, c4 = (idx & 31) * 4;
        *(float4*)&out[(size_t)(base + row) * 128 + c4]   = *(const float4*)&ot[row * 128 + c4];
        *(float4*)&zpack[(size_t)(base + row) * 128 + c4] = *(const float4*)&zt[row * 128 + c4];
    }
}

// ---------------------------------------------------------------------------
// Kernel 2: fused edge pipeline. 4 waves/block, 16 edges/wave, 24 WMMAs/wave.
//   h = silu(ef @ fc_w1 / 4), w = h @ fc_w2 / 8, then 5 TP paths + atomics.
// ---------------------------------------------------------------------------
__global__ void __launch_bounds__(128, 1)
edge_conv(const int* __restrict__ edge_src,
          const int* __restrict__ edge_dst,
          const float* __restrict__ edge_attr,
          const float* __restrict__ edge_features,
          const _Float16* __restrict__ wf,
          const float* __restrict__ zpack,
          float* __restrict__ g0, float* __restrict__ g1,
          int nE)
{
    __shared__ __align__(16) float efs[4][16 * NEF];    //  4 KB
    __shared__ __align__(16) float hbuf[4][16 * RH];    // 16 KB
    __shared__ __align__(16) float wbuf[4][16 * WNUM];  // 40 KB
    const int wave = threadIdx.x >> 5;
    const int lane = threadIdx.x & 31;
    int e0 = (blockIdx.x * 4 + wave) * 16;
    const bool active = (e0 + 16 <= nE);
    if (e0 + 16 > nE) e0 = (nE >= 16) ? nE - 16 : 0;

    // stage edge_features tile: 64 float4, 2 per lane (coalesced)
#pragma unroll
    for (int j = 0; j < 2; ++j) {
        const int idx = j * 32 + lane;
        const int row = idx >> 2, c4 = (idx & 3) * 4;
        *(float4*)&efs[wave][row * NEF + c4] =
            *(const float4*)&edge_features[(size_t)(e0 + row) * NEF + c4];
    }

    const int kb = (lane >> 4) << 3;
    const int col0 = lane & 15;

    // ---- GEMM 1: ef (16x16 pad->32) @ fc_w1 (16x64) ----
    v16h aef;
    {
        const int r0 = (lane & 15) * NEF;
        cvt8(aef, 0, *(const float4*)&efs[wave][r0 + kb],
                      *(const float4*)&efs[wave][r0 + kb + 4]);
#pragma unroll
        for (int i = 8; i < 16; ++i) aef[i] = (_Float16)0.0f;  // K pad
    }
#pragma unroll 1
    for (int ct = 0; ct < 4; ++ct) {
        v8f c = {};
        c = wmma16(aef, ldfrag(wf, FR_FC1 + ct, lane), c);
#pragma unroll
        for (int r = 0; r < 8; ++r)
            hbuf[wave][d_row(lane, r) * RH + ct * 16 + col0] = silu(c[r] * INV_S16);
    }
    __syncthreads();

    // ---- GEMM 2: h (16x64) @ fc_w2 (64x160), rolled t loop (no spills) ----
    v16h ah0, ah1;
    {
        const int rh = (lane & 15) * RH;
        cvt8(ah0, 0, *(const float4*)&hbuf[wave][rh + kb],      *(const float4*)&hbuf[wave][rh + kb + 4]);
        cvt8(ah0, 8, *(const float4*)&hbuf[wave][rh + 16 + kb], *(const float4*)&hbuf[wave][rh + 16 + kb + 4]);
        cvt8(ah1, 0, *(const float4*)&hbuf[wave][rh + 32 + kb], *(const float4*)&hbuf[wave][rh + 32 + kb + 4]);
        cvt8(ah1, 8, *(const float4*)&hbuf[wave][rh + 48 + kb], *(const float4*)&hbuf[wave][rh + 48 + kb + 4]);
    }
#pragma unroll 1
    for (int t = 0; t < 10; ++t) {
        v8f c = {};
        c = wmma16(ah0, ldfrag(wf, FR_FC2 + t,      lane), c);
        c = wmma16(ah1, ldfrag(wf, FR_FC2 + 10 + t, lane), c);
#pragma unroll
        for (int r = 0; r < 8; ++r)
            wbuf[wave][d_row(lane, r) * WNUM + t * 16 + col0] = c[r] * INV_S64;
    }
    __syncthreads();

    if (!active) return;   // uniform per wave; after all barriers

    // ---- tensor-product paths + scatter: edge uniform per iter, u = lane ----
#pragma unroll 2
    for (int e = 0; e < 16; ++e) {
        const int ge = e0 + e;                       // wave-uniform
        const int src = edge_src[ge];                // scalar-promotable
        const int dst = edge_dst[ge];
        const float4 ea = *(const float4*)&edge_attr[(size_t)ge * 4];  // a0,a1x,a1y,a1z
        const float4 zv = *(const float4*)&zpack[(size_t)src * 128 + lane * 4]; // coalesced
        const float zz0 = zv.x, z1x = zv.y, z1y = zv.z, z1z = zv.w;
        const float* we = &wbuf[wave][e * WNUM];
        const float w0p = we[lane];
        const float w1p = we[MUL + lane];
        const float w2p = we[2 * MUL + lane];
        const float w3p = we[3 * MUL + lane];
        const float w4p = we[4 * MUL + lane];

        // path 0: y0a = w0 * z0 * a0
        atomicAdd(&g0[(size_t)dst * 64 + lane], w0p * zz0 * ea.x * INV_NN);
        // path 3: y0b = w3 * (z1 . a1) / sqrt(3)
        const float dot = z1x * ea.y + z1y * ea.z + z1z * ea.w;
        atomicAdd(&g0[(size_t)dst * 64 + 32 + lane], w3p * dot * (INV_SQRT3 * INV_NN));
        // path 1: y1a = (w1 * z0) * a1[m]
        const float t1 = w1p * zz0 * INV_NN;
        atomicAdd(&g1[(size_t)dst * 288 + lane * 3 + 0], t1 * ea.y);
        atomicAdd(&g1[(size_t)dst * 288 + lane * 3 + 1], t1 * ea.z);
        atomicAdd(&g1[(size_t)dst * 288 + lane * 3 + 2], t1 * ea.w);
        // path 2: y1b = (w2 * a0) * z1[m]
        const float t2 = w2p * ea.x * INV_NN;
        atomicAdd(&g1[(size_t)dst * 288 + (MUL + lane) * 3 + 0], t2 * z1x);
        atomicAdd(&g1[(size_t)dst * 288 + (MUL + lane) * 3 + 1], t2 * z1y);
        atomicAdd(&g1[(size_t)dst * 288 + (MUL + lane) * 3 + 2], t2 * z1z);
        // path 4: y1c = w4 * cross(z1, a1) / sqrt(2)
        const float t4 = w4p * (INV_SQRT2 * INV_NN);
        atomicAdd(&g1[(size_t)dst * 288 + (2 * MUL + lane) * 3 + 0], t4 * (z1y * ea.w - z1z * ea.z));
        atomicAdd(&g1[(size_t)dst * 288 + (2 * MUL + lane) * 3 + 1], t4 * (z1z * ea.y - z1x * ea.w));
        atomicAdd(&g1[(size_t)dst * 288 + (2 * MUL + lane) * 3 + 2], t4 * (z1x * ea.z - z1y * ea.y));
    }
}

// ---------------------------------------------------------------------------
// Kernel 3: lin2 fctp on aggregated g; out += C_X * o (C_S*s already there).
// One wave = 16 nodes, 22 WMMAs. o staged in LDS, vectorized RMW on out.
// ---------------------------------------------------------------------------
__global__ void __launch_bounds__(32, 1)
node_out(const float* __restrict__ node_attr,
         const float* __restrict__ g0,
         const float* __restrict__ g1,
         const _Float16* __restrict__ wf,
         float* __restrict__ out, int n)
{
    __shared__ __align__(16) float gs[16 * 352];   // [row][0:64]=g0, [row][64:352]=g1
    __shared__ __align__(16) float ot[16 * 128];   // C_X*o tile
    __shared__ float av[16];
    const int lane = threadIdx.x & 31;
    int base = blockIdx.x * 16;
    if (base + 16 > n) base = n - 16;

#pragma unroll
    for (int j = 0; j < 8; ++j) {
        const int idx = j * 32 + lane;
        const int row = idx >> 4, c4 = (idx & 15) * 4;
        *(float4*)&gs[row * 352 + c4] =
            *(const float4*)&g0[(size_t)(base + row) * 64 + c4];
    }
#pragma unroll
    for (int j = 0; j < 36; ++j) {
        const int idx = j * 32 + lane;
        const int row = idx / 72, c4 = (idx % 72) * 4;
        *(float4*)&gs[row * 352 + 64 + c4] =
            *(const float4*)&g1[(size_t)(base + row) * 288 + c4];
    }
    if (lane < 16) av[lane] = node_attr[base + lane];
    __syncthreads();

    float af0[8], af1[8];
#pragma unroll
    for (int r = 0; r < 8; ++r) {
        const float a = av[d_row(lane, r)];
        af0[r] = a * (INV_S64 * C_X);
        af1[r] = a * (INV_S96 * C_X);
    }

    const int r0 = (lane & 15) * 352;
    const int kb = (lane >> 4) << 3;
    const int col0 = lane & 15;

    // ---- o0: K=64 ----
    v16h ag0, ag1;
    cvt8(ag0, 0, *(const float4*)&gs[r0 + kb],      *(const float4*)&gs[r0 + kb + 4]);
    cvt8(ag0, 8, *(const float4*)&gs[r0 + 16 + kb], *(const float4*)&gs[r0 + 16 + kb + 4]);
    cvt8(ag1, 0, *(const float4*)&gs[r0 + 32 + kb], *(const float4*)&gs[r0 + 32 + kb + 4]);
    cvt8(ag1, 8, *(const float4*)&gs[r0 + 48 + kb], *(const float4*)&gs[r0 + 48 + kb + 4]);
#pragma unroll 1
    for (int half = 0; half < 2; ++half) {
        v8f c = {};
        c = wmma16(ag0, ldfrag(wf, FR_L20 + half,     lane), c);
        c = wmma16(ag1, ldfrag(wf, FR_L20 + 2 + half, lane), c);
        const int col = half * 16 + col0;
#pragma unroll
        for (int r = 0; r < 8; ++r)
            ot[d_row(lane, r) * 128 + col] = c[r] * af0[r];
    }

    // ---- o1: K=96 per m (rolled) ----
#pragma unroll 1
    for (int m = 0; m < 3; ++m) {
        v16h am0, am1, am2;
#pragma unroll
        for (int i = 0; i < 16; ++i) {
            am0[i] = (_Float16)gs[r0 + 64 + a_k(lane, i) * 3 + m];
            am1[i] = (_Float16)gs[r0 + 64 + (32 + a_k(lane, i)) * 3 + m];
            am2[i] = (_Float16)gs[r0 + 64 + (64 + a_k(lane, i)) * 3 + m];
        }
#pragma unroll 1
        for (int half = 0; half < 2; ++half) {
            v8f c = {};
            c = wmma16(am0, ldfrag(wf, FR_L21 + half,     lane), c);
            c = wmma16(am1, ldfrag(wf, FR_L21 + 2 + half, lane), c);
            c = wmma16(am2, ldfrag(wf, FR_L21 + 4 + half, lane), c);
            const int col = half * 16 + col0;
#pragma unroll
            for (int r = 0; r < 8; ++r)
                ot[d_row(lane, r) * 128 + 32 + col * 3 + m] = c[r] * af1[r];
        }
    }
    __syncthreads();

    // vectorized read-modify-write: out += ot
#pragma unroll
    for (int j = 0; j < 16; ++j) {
        const int idx = j * 32 + lane;
        const int row = idx >> 5, c4 = (idx & 31) * 4;
        float* op = &out[(size_t)(base + row) * 128 + c4];
        float4 prev = *(const float4*)op;
        const float4 ov = *(const float4*)&ot[row * 128 + c4];
        prev.x += ov.x; prev.y += ov.y; prev.z += ov.z; prev.w += ov.w;
        *(float4*)op = prev;
    }
}

// ---------------------------------------------------------------------------
extern "C" void kernel_launch(void* const* d_in, const int* in_sizes, int n_in,
                              void* d_out, int out_size, void* d_ws, size_t ws_size,
                              hipStream_t stream) {
    const float* node_input    = (const float*)d_in[0];
    const float* node_attr     = (const float*)d_in[1];
    const int*   edge_src      = (const int*)  d_in[2];
    const int*   edge_dst      = (const int*)  d_in[3];
    const float* edge_attr     = (const float*)d_in[4];
    const float* edge_features = (const float*)d_in[5];
    const float* fc_w1         = (const float*)d_in[6];
    const float* fc_w2         = (const float*)d_in[7];
    const float* sc_w0         = (const float*)d_in[8];
    const float* sc_w1         = (const float*)d_in[9];
    const float* l1_w0         = (const float*)d_in[10];
    const float* l1_w1         = (const float*)d_in[11];
    const float* l2_w0         = (const float*)d_in[12];
    const float* l2_w1         = (const float*)d_in[13];

    const int n = in_sizes[1];   // N (node_attr)
    const int e = in_sizes[2];   // E (edge_src)

    // ws layout (floats): zpack (N*128) | g0 (N*64) | g1 (N*288) | wf16
    float* ws = (float*)d_ws;
    float* zpack = ws;
    float* g0 = ws + (size_t)n * 128;
    float* g1 = ws + (size_t)n * 192;
    _Float16* wf = (_Float16*)(ws + (size_t)n * 480);

    prep_weights<<<1, 256, 0, stream>>>(sc_w0, sc_w1, l1_w0, l1_w1,
                                        fc_w1, fc_w2, l2_w0, l2_w1, wf);

    zero_g<<<2048, 256, 0, stream>>>((float4*)g0, (long long)n * 88);

    node_lin<<<(n + 15) / 16, 32, 0, stream>>>(
        node_input, node_attr, wf, zpack, (float*)d_out, n);

    edge_conv<<<(e + 63) / 64, 128, 0, stream>>>(
        edge_src, edge_dst, edge_attr, edge_features, wf,
        zpack, g0, g1, e);

    node_out<<<(n + 15) / 16, 32, 0, stream>>>(
        node_attr, g0, g1, wf, (float*)d_out, n);
}
